// GCN_18236431139008
// MI455X (gfx1250) — matrix operand order repaired and checked by reading, not compiled, not run on hardware
//
#include <hip/hip_runtime.h>
#include <hip/hip_bf16.h>
#include <stdint.h>
#include <stddef.h>

// ---------------------------------------------------------------------------
// Problem constants (from reference)
// ---------------------------------------------------------------------------
#define NNODES 50000
#define NEDGES 800000
#define FDIM   768
#define HDIM   256
#define NGRAPH 4
#define NB     32
#define C1DIM  1280   // H*(G+1)
#define C2DIM  640    // C1/2

typedef __bf16 bfv4  __attribute__((ext_vector_type(4)));
typedef __bf16 bfv8  __attribute__((ext_vector_type(8)));
typedef __bf16 bfv16 __attribute__((ext_vector_type(16)));
typedef float  v8f   __attribute__((ext_vector_type(8)));

// ---------------------------------------------------------------------------
// Elementwise helpers (float4 / bf16x4 vectorized)
// ---------------------------------------------------------------------------
__global__ void k_f32_to_bf16(const float* __restrict__ in, __bf16* __restrict__ out, long n4) {
  long i  = (long)blockIdx.x * blockDim.x + threadIdx.x;
  long st = (long)gridDim.x * blockDim.x;
  for (; i < n4; i += st) {
    float4 v = ((const float4*)in)[i];
    bfv4 o = { (__bf16)v.x, (__bf16)v.y, (__bf16)v.z, (__bf16)v.w };
    ((bfv4*)out)[i] = o;
  }
}

// convert + transpose: in = [K x N] f32 row-major, out = [N x K] bf16 row-major
__global__ void k_f32_to_bf16_tr(const float* __restrict__ in, __bf16* __restrict__ out,
                                 int K, int N) {
  int idx = blockIdx.x * blockDim.x + threadIdx.x;
  if (idx < K * N) {
    int k = idx / N, n = idx % N;            // coalesced read over n
    out[(long)n * K + k] = (__bf16)in[idx];
  }
}

__global__ void k_zero_f32(float* __restrict__ p, long n4) {
  long i  = (long)blockIdx.x * blockDim.x + threadIdx.x;
  long st = (long)gridDim.x * blockDim.x;
  float4 z = {0.f, 0.f, 0.f, 0.f};
  for (; i < n4; i += st) ((float4*)p)[i] = z;
}

// ---------------------------------------------------------------------------
// Degree / symmetric normalization   deg[d] = indegree + 1 (self loop)
// ---------------------------------------------------------------------------
__global__ void k_degree(const int* __restrict__ dst, float* __restrict__ deg, int e) {
  int i = blockIdx.x * blockDim.x + threadIdx.x;
  if (i < e) atomicAdd(&deg[dst[i]], 1.0f);
}

__global__ void k_dinv(const float* __restrict__ deg, float* __restrict__ dinv, int n) {
  int i = blockIdx.x * blockDim.x + threadIdx.x;
  if (i < n) dinv[i] = rsqrtf(deg[i] + 1.0f);   // always > 0 due to self loop
}

// ---------------------------------------------------------------------------
// Edge scatter: out[d,:] += h[s,:] * dinv[s]*dinv[d]
// 4 edges / 256-thread block; 64 threads x float4 per edge. N*H operand is
// 51.2 MB -> resident in the 192 MB L2; gathers and f32 atomics stay on-chip.
// ---------------------------------------------------------------------------
__global__ __launch_bounds__(256)
void k_edge_scatter(const float* __restrict__ h, const int* __restrict__ src,
                    const int* __restrict__ dst, const float* __restrict__ dinv,
                    float* __restrict__ out, int E) {
  int e = blockIdx.x * 4 + (threadIdx.x >> 6);
  if (e >= E) return;
  int s = src[e], d = dst[e];
  float w = dinv[s] * dinv[d];
  int c = (threadIdx.x & 63) * 4;
  float4 hv = *(const float4*)&h[(long)s * HDIM + c];
  float* o  = &out[(long)d * HDIM + c];
  atomicAdd(o + 0, hv.x * w);
  atomicAdd(o + 1, hv.y * w);
  atomicAdd(o + 2, hv.z * w);
  atomicAdd(o + 3, hv.w * w);
}

// agg = [relu](agg + dinv[i]^2 * h + bias); also emit bf16 copy for next GEMM
__global__ __launch_bounds__(256)
void k_finalize(float* __restrict__ agg, const float* __restrict__ h,
                const float* __restrict__ dinv, const float* __restrict__ bias,
                __bf16* __restrict__ obf, int relu, int n) {
  int idx = blockIdx.x * blockDim.x + threadIdx.x;   // node*64 + cquad
  long i = idx >> 6; int c = (idx & 63) * 4;
  if (i >= n) return;
  float di = dinv[i]; float w = di * di;
  float4 av = *(float4*)&agg[i * HDIM + c];
  float4 hv = *(const float4*)&h[i * HDIM + c];
  float4 bv = *(const float4*)&bias[c];
  float4 v  = { av.x + w * hv.x + bv.x, av.y + w * hv.y + bv.y,
                av.z + w * hv.z + bv.z, av.w + w * hv.w + bv.w };
  if (relu) {
    v.x = fmaxf(v.x, 0.f); v.y = fmaxf(v.y, 0.f);
    v.z = fmaxf(v.z, 0.f); v.w = fmaxf(v.w, 0.f);
  }
  *(float4*)&agg[i * HDIM + c] = v;
  if (obf) {
    bfv4 o = { (__bf16)v.x, (__bf16)v.y, (__bf16)v.z, (__bf16)v.w };
    *(bfv4*)&obf[i * HDIM + c] = o;
  }
}

// ---------------------------------------------------------------------------
// Global mean pool (segment mean over batch ids)
// ---------------------------------------------------------------------------
__global__ __launch_bounds__(256)
void k_pool_accum(const float* __restrict__ h, const int* __restrict__ batch,
                  float* __restrict__ pooled, int n) {
  int idx = blockIdx.x * blockDim.x + threadIdx.x;
  long i = idx >> 6; int c = (idx & 63) * 4;
  if (i >= n) return;
  int b = batch[i];
  float4 hv = *(const float4*)&h[i * HDIM + c];
  float* o = &pooled[(long)b * HDIM + c];
  atomicAdd(o + 0, hv.x); atomicAdd(o + 1, hv.y);
  atomicAdd(o + 2, hv.z); atomicAdd(o + 3, hv.w);
}

__global__ void k_pool_count(const int* __restrict__ batch, float* __restrict__ cnt, int n) {
  int i = blockIdx.x * blockDim.x + threadIdx.x;
  if (i < n) atomicAdd(&cnt[batch[i]], 1.0f);
}

__global__ void k_pool_write(const float* __restrict__ pooled, const float* __restrict__ cnt,
                             float* __restrict__ out, int ldo) {
  int idx = blockIdx.x * blockDim.x + threadIdx.x;   // NB*HDIM threads
  int b = idx >> 8, c = idx & 255;
  out[(long)b * ldo + c] = pooled[idx] / fmaxf(cnt[b], 1.0f);
}

// ---------------------------------------------------------------------------
// bf16 WMMA GEMM:  C[M x Nout] = A[M x K] * Bt[Nout x K]^T  (+bias, +relu)
//
// Bt is pre-transposed (output-channel-major, K contiguous) so the per-block
// B tile (64 rows x K) is a single contiguous region: staged into LDS ONCE
// per block with b128 copies, zero barriers in the K loop.
//
// Block = 256 threads = 8 wave32 waves; each wave computes a 32x64 tile
// (8 f32 accumulators). Per 32-K step: 4 global b128 (two A frags),
// 8 ds_load_b128 (4 B frags, each reused by both row tiles), 8 WMMA.
// ---------------------------------------------------------------------------
__device__ inline bfv16 load_frag16(const __bf16* p) {
  bfv8 lo = *(const bfv8*)(p);
  bfv8 hi = *(const bfv8*)(p + 16);
  return __builtin_shufflevector(lo, hi,
      0, 1, 2, 3, 4, 5, 6, 7, 8, 9, 10, 11, 12, 13, 14, 15);
}

__global__ __launch_bounds__(256)
void k_wmma_gemm(const __bf16* __restrict__ A, int lda,
                 const __bf16* __restrict__ Bt, int K,
                 float* __restrict__ C, int ldc,
                 int M,
                 const float* __restrict__ bias, int relu) {
  extern __shared__ __bf16 Bs[];            // 64 * K halfs

  const int tid   = threadIdx.x;
  const int lane  = tid & 31;
  const int wave  = tid >> 5;               // 0..7
  const int mBase = blockIdx.x * 256 + wave * 32;
  const int nBase = blockIdx.y * 64;

  // stage contiguous 64 x K B tile into LDS (one time)
  {
    const bfv8* sp = (const bfv8*)(Bt + (long)nBase * K);
    bfv8* dp = (bfv8*)Bs;
    const int nchunk = (64 * K) >> 3;
    for (int idx = tid; idx < nchunk; idx += 256) dp[idx] = sp[idx];
  }
  __syncthreads();

  int row0 = mBase + (lane & 15);
  int row1 = mBase + 16 + (lane & 15);
  if (row0 >= M) row0 = M - 1;              // clamp loads; stores guarded below
  if (row1 >= M) row1 = M - 1;
  const int kOffA = (lane >> 4) * 8;        // A 16-bit interleave: 0 or 8
  const int kOffB = (lane >> 4) * 8;        // B 16-bit interleave: 0 or 8 (same as A)
  const int bRow  = lane & 15;

  v8f acc[2][4] = {};

  for (int k0 = 0; k0 < K; k0 += 32) {
    bfv16 a0 = load_frag16(A + (long)row0 * lda + k0 + kOffA);
    bfv16 a1 = load_frag16(A + (long)row1 * lda + k0 + kOffA);
#pragma unroll
    for (int j = 0; j < 4; ++j) {
      bfv16 bf = load_frag16(&Bs[(long)(j * 16 + bRow) * K + k0 + kOffB]);
      acc[0][j] = __builtin_amdgcn_wmma_f32_16x16x32_bf16(
          false, a0, false, bf, (short)0, acc[0][j], false, false);
      acc[1][j] = __builtin_amdgcn_wmma_f32_16x16x32_bf16(
          false, a1, false, bf, (short)0, acc[1][j], false, false);
    }
  }

  // C/D layout: VGPR r -> row r (+8 for lanes 16-31), col = lane&15
  const int rOff = (lane >> 4) * 8;
#pragma unroll
  for (int half = 0; half < 2; ++half) {
#pragma unroll
    for (int j = 0; j < 4; ++j) {
      const int col = nBase + j * 16 + (lane & 15);
      float badd = bias ? bias[col] : 0.0f;
#pragma unroll
      for (int r = 0; r < 8; ++r) {
        int row = mBase + half * 16 + rOff + r;
        if (row < M) {
          float v = acc[half][j][r] + badd;
          if (relu) v = fmaxf(v, 0.0f);
          C[(long)row * ldc + col] = v;
        }
      }
    }
  }
}

// ---------------------------------------------------------------------------
// Final tiny FC: logits[32,2] = hid[32,640] @ Wl2[640,2] + bl2
// ---------------------------------------------------------------------------
__global__ void k_final_fc(const float* __restrict__ hid, const float* __restrict__ W,
                           const float* __restrict__ b, float* __restrict__ out) {
  int t  = threadIdx.x;        // 64 threads
  int bi = t >> 1, j = t & 1;
  float acc = b[j];
  for (int k = 0; k < C2DIM; ++k) acc += hid[bi * C2DIM + k] * W[k * 2 + j];
  out[bi * 2 + j] = acc;
}

// ---------------------------------------------------------------------------
// Launch
// ---------------------------------------------------------------------------
extern "C" void kernel_launch(void* const* d_in, const int* in_sizes, int n_in,
                              void* d_out, int out_size, void* d_ws, size_t ws_size,
                              hipStream_t stream) {
  (void)in_sizes; (void)n_in; (void)out_size; (void)ws_size;

  const float* x_g[NGRAPH]; const int* ei[NGRAPH]; const int* batch[NGRAPH];
  for (int g = 0; g < NGRAPH; ++g) {
    x_g[g]   = (const float*)d_in[3 * g + 0];
    ei[g]    = (const int*)  d_in[3 * g + 1];   // [2,E] flat: src = +0, dst = +E
    batch[g] = (const int*)  d_in[3 * g + 2];
  }
  const float* x_code = (const float*)d_in[12];
  const float* W1  = (const float*)d_in[13]; const float* b1  = (const float*)d_in[14];
  const float* W2  = (const float*)d_in[15]; const float* b2  = (const float*)d_in[16];
  const float* W3  = (const float*)d_in[17]; const float* b3  = (const float*)d_in[18];
  const float* Wc  = (const float*)d_in[19]; const float* bc  = (const float*)d_in[20];
  const float* Wl1 = (const float*)d_in[21]; const float* bl1 = (const float*)d_in[22];
  const float* Wl2 = (const float*)d_in[23]; const float* bl2 = (const float*)d_in[24];
  float* out = (float*)d_out;

  // workspace carve-up (~208 MB total), 256B aligned slices
  char* p = (char*)d_ws;
  auto take = [&](size_t bytes) -> char* {
    char* r = p; p += (bytes + 255) & ~(size_t)255; return r;
  };
  __bf16* Abf    = (__bf16*)take((size_t)NNODES * FDIM * 2);   // 76.8 MB
  float*  Cf     = (float*) take((size_t)NNODES * HDIM * 4);   // 51.2 MB
  float*  Hf     = (float*) take((size_t)NNODES * HDIM * 4);   // 51.2 MB
  __bf16* Hbf    = (__bf16*)take((size_t)NNODES * HDIM * 2);   // 25.6 MB
  float*  deg    = (float*) take((size_t)NNODES * 4);
  float*  dinv   = (float*) take((size_t)NNODES * 4);
  __bf16* W1t    = (__bf16*)take((size_t)FDIM * HDIM * 2);     // transposed [H x F]
  __bf16* W2t    = (__bf16*)take((size_t)HDIM * HDIM * 2);
  __bf16* W3t    = (__bf16*)take((size_t)HDIM * HDIM * 2);
  __bf16* Wct    = (__bf16*)take((size_t)FDIM * HDIM * 2);
  __bf16* Wl1t   = (__bf16*)take((size_t)C1DIM * C2DIM * 2);   // [640 x 1280]
  __bf16* xcb    = (__bf16*)take((size_t)NB * FDIM * 2);
  float*  pooled = (float*) take((size_t)NB * HDIM * 4);
  float*  cnt    = (float*) take((size_t)NB * 4);
  float*  conc   = (float*) take((size_t)NB * C1DIM * 4);
  __bf16* concb  = (__bf16*)take((size_t)NB * C1DIM * 2);
  float*  hid    = (float*) take((size_t)NB * C2DIM * 4);

  auto cvt = [&](const float* s, __bf16* d, long n) {
    long n4 = n / 4;
    int blocks = (int)((n4 + 255) / 256); if (blocks > 4096) blocks = 4096;
    hipLaunchKernelGGL(k_f32_to_bf16, dim3(blocks), dim3(256), 0, stream, s, d, n4);
  };
  auto cvt_tr = [&](const float* s, __bf16* d, int K, int N) {
    int blocks = (K * N + 255) / 256;
    hipLaunchKernelGGL(k_f32_to_bf16_tr, dim3(blocks), dim3(256), 0, stream, s, d, K, N);
  };
  auto zero = [&](float* ptr, long n) {
    long n4 = n / 4;
    int blocks = (int)((n4 + 255) / 256); if (blocks > 4096) blocks = 4096;
    hipLaunchKernelGGL(k_zero_f32, dim3(blocks), dim3(256), 0, stream, ptr, n4);
  };
  auto gemm = [&](const __bf16* A, int lda, const __bf16* Bt, int K,
                  float* C, int ldc, int M, int Nout,
                  const float* bias, int relu) {
    dim3 grid((unsigned)((M + 255) / 256), (unsigned)(Nout / 64));
    size_t shmem = (size_t)64 * K * 2;
    hipLaunchKernelGGL(k_wmma_gemm, grid, dim3(256), shmem, stream,
                       A, lda, Bt, K, C, ldc, M, bias, relu);
  };

  // per-call weight conversions (bf16, transposed to [N x K])
  cvt_tr(W1,  W1t,  FDIM,  HDIM);
  cvt_tr(W2,  W2t,  HDIM,  HDIM);
  cvt_tr(W3,  W3t,  HDIM,  HDIM);
  cvt_tr(Wc,  Wct,  FDIM,  HDIM);
  cvt_tr(Wl1, Wl1t, C1DIM, C2DIM);
  cvt(x_code, xcb, (long)NB * FDIM);

  for (int g = 0; g < NGRAPH; ++g) {
    const int* src = ei[g];
    const int* dst = ei[g] + NEDGES;

    cvt(x_g[g], Abf, (long)NNODES * FDIM);

    zero(deg, NNODES);
    hipLaunchKernelGGL(k_degree, dim3((NEDGES + 255) / 256), dim3(256), 0, stream,
                       dst, deg, NEDGES);
    hipLaunchKernelGGL(k_dinv, dim3((NNODES + 255) / 256), dim3(256), 0, stream,
                       deg, dinv, NNODES);

    const int nFin = (NNODES * 64 + 255) / 256;

    // conv1: h = relu(norm_agg(X @ W1) + b1)
    gemm(Abf, FDIM, W1t, FDIM, Cf, HDIM, NNODES, HDIM, nullptr, 0);
    zero(Hf, (long)NNODES * HDIM);
    hipLaunchKernelGGL(k_edge_scatter, dim3(NEDGES / 4), dim3(256), 0, stream,
                       Cf, src, dst, dinv, Hf, NEDGES);
    hipLaunchKernelGGL(k_finalize, dim3(nFin), dim3(256), 0, stream,
                       Hf, Cf, dinv, b1, Hbf, 1, NNODES);

    // conv2
    gemm(Hbf, HDIM, W2t, HDIM, Cf, HDIM, NNODES, HDIM, nullptr, 0);
    zero(Hf, (long)NNODES * HDIM);
    hipLaunchKernelGGL(k_edge_scatter, dim3(NEDGES / 4), dim3(256), 0, stream,
                       Cf, src, dst, dinv, Hf, NEDGES);
    hipLaunchKernelGGL(k_finalize, dim3(nFin), dim3(256), 0, stream,
                       Hf, Cf, dinv, b2, Hbf, 1, NNODES);

    // conv3 (no relu)
    gemm(Hbf, HDIM, W3t, HDIM, Cf, HDIM, NNODES, HDIM, nullptr, 0);
    zero(Hf, (long)NNODES * HDIM);
    hipLaunchKernelGGL(k_edge_scatter, dim3(NEDGES / 4), dim3(256), 0, stream,
                       Cf, src, dst, dinv, Hf, NEDGES);
    hipLaunchKernelGGL(k_finalize, dim3(nFin), dim3(256), 0, stream,
                       Hf, Cf, dinv, b3, (__bf16*)nullptr, 0, NNODES);

    // global mean pool into concat column block g
    zero(pooled, (long)NB * HDIM);
    zero(cnt, NB);
    hipLaunchKernelGGL(k_pool_accum, dim3(nFin), dim3(256), 0, stream,
                       Hf, batch[g], pooled, NNODES);
    hipLaunchKernelGGL(k_pool_count, dim3((NNODES + 255) / 256), dim3(256), 0, stream,
                       batch[g], cnt, NNODES);
    hipLaunchKernelGGL(k_pool_write, dim3((NB * HDIM) / 256), dim3(256), 0, stream,
                       pooled, cnt, conc + g * HDIM, C1DIM);
  }

  // x_code branch -> concat columns [4H .. 5H)
  gemm(xcb, FDIM, Wct, FDIM, conc + NGRAPH * HDIM, C1DIM, NB, HDIM, bc, 1);

  // MLP head
  cvt(conc, concb, (long)NB * C1DIM);
  gemm(concb, C1DIM, Wl1t, C1DIM, hid, C2DIM, NB, C2DIM, bl1, 1);
  hipLaunchKernelGGL(k_final_fc, dim3(1), dim3(64), 0, stream, hid, Wl2, bl2, out);
}
